// MViTV2SelfAttention_51402168599292
// MI455X (gfx1250) — matrix operand order repaired and checked by reading, not compiled, not run on hardware
//
#include <hip/hip_runtime.h>

typedef _Float16 v8h  __attribute__((ext_vector_type(8)));
typedef _Float16 v16h __attribute__((ext_vector_type(16)));
typedef float    v8f  __attribute__((ext_vector_type(8)));
typedef unsigned int u32x4 __attribute__((ext_vector_type(4)));
typedef int          i32x4 __attribute__((ext_vector_type(4)));
typedef int          i32x8 __attribute__((ext_vector_type(8)));

union U16 { v16h v; v8h h8[2]; };

#define NB     16      // batch
#define NH     4       // heads
#define HD     96      // head dim
#define NTOK   3136    // 56*56
#define H_IN   56
#define W_IN   56
#define OHW    28
#define NPOOL  784     // 28*28
#define NPAD   800     // padded token count (multiple of 32)
#define IND    192
#define QKVD   1152
#define OUTD   384
#define NBH    64      // NB*NH
#define SCALE_F 0.10206207261596577f  // 96^-0.5

__device__ inline v8f v8f_zero() {
  v8f r;
#pragma unroll
  for (int i = 0; i < 8; ++i) r[i] = 0.0f;
  return r;
}

__device__ inline v8f wmma_f16(v16h a, v16h b, v8f c) {
  // D = A(16x32 f16) * B(32x16 f16) + C(16x16 f32)
  return __builtin_amdgcn_wmma_f32_16x16x32_f16(false, a, false, b,
                                                (short)0, c, false, false);
}

// ---------------------------------------------------------------------------
// TDM: 2-D tile load Global -> LDS via tensor descriptor (ISA 08, D# groups).
// tile_w elements per row, tile_h rows, row stride in elements, 4-byte data.
// Descriptor packing per CDNA5 ISA sec 8.3/8.4:
//   g0: [1:0]=count=1, [63:32]=lds_addr, [120:64]=global_addr, [127:126]=2
//   g1: [17:16]=data_size, [79:48]=tensor_dim0, [111:80]=tensor_dim1,
//       [127:112]=tile_dim0, [143:128]=tile_dim1, [207:160]=dim0_stride
//   g2/g3: zero (2-D tensor)
// ---------------------------------------------------------------------------
__device__ inline void tdm_load_tile_f32(unsigned lds_byte_off, const void* gptr,
                                         unsigned tile_w, unsigned tile_h,
                                         unsigned row_stride_elems)
{
  const unsigned long long ga = (unsigned long long)(size_t)gptr;
  const unsigned data_size_code = 2u;   // 4-byte elements
  u32x4 g0 = { 1u,
               lds_byte_off,
               (unsigned)ga,
               (unsigned)((ga >> 32) & 0x01ffffffull) | 0x80000000u };
  i32x8 g1 = { (int)(data_size_code << 16),
               (int)((tile_w & 0xffffu) << 16),                 // tensor_dim0 lo
               (int)((tile_w >> 16) | ((tile_h & 0xffffu) << 16)), // d0 hi, d1 lo
               (int)((tile_h >> 16) | (tile_w << 16)),          // d1 hi, tile_dim0
               (int)(tile_h & 0xffffu),                         // tile_dim1
               (int)row_stride_elems,                           // dim0_stride lo32
               0, 0 };
  i32x4 gz = { 0, 0, 0, 0 };
#if defined(__clang_major__) && (__clang_major__ >= 23)
  i32x8 gz8 = { 0, 0, 0, 0, 0, 0, 0, 0 };
  __builtin_amdgcn_tensor_load_to_lds(g0, g1, gz, gz, gz8, 0);
#else
  __builtin_amdgcn_tensor_load_to_lds(g0, g1, gz, gz, 0);
#endif
}

// ---------------------------------------------------------------------------
// Kernel 1: QKV projection GEMM  [50176 x 192] x [192 x 1152] + bias
// A tile staged by the Tensor Data Mover (async DMA, TENSORcnt);
// B tile staged f32->f16 by the threads. Output scattered as f16.
// ---------------------------------------------------------------------------
__global__ __launch_bounds__(256) void qkv_gemm_kernel(
    const float* __restrict__ hs, const float* __restrict__ w,
    const float* __restrict__ bias,
    _Float16* __restrict__ qraw, _Float16* __restrict__ kraw,
    _Float16* __restrict__ vraw)
{
  __shared__ __attribute__((aligned(16))) float    Ashf[128 * 32];  // TDM dest
  __shared__ __attribute__((aligned(16))) _Float16 Bsh[64 * 40];

  const int tid  = threadIdx.x;
  const int lane = tid & 31;
  const int wv   = tid >> 5;          // 8 waves
  const int m0   = blockIdx.x * 128;
  const int n0   = blockIdx.y * 64;
  const int g    = lane >> 4;
  const int ml   = lane & 15;

  v8f acc[4];
#pragma unroll
  for (int ns = 0; ns < 4; ++ns) acc[ns] = v8f_zero();

  for (int k0 = 0; k0 < IND; k0 += 32) {
    __syncthreads();   // previous iteration's fragment reads are done

    // A tile 128x32 f32 via Tensor Data Mover (wave 0 issues, EXEC ignored)
    if (wv == 0) {
      tdm_load_tile_f32((unsigned)(size_t)(void*)Ashf,
                        hs + (size_t)m0 * IND + k0,
                        /*tile_w=*/32, /*tile_h=*/128,
                        /*row_stride=*/IND);
      __builtin_amdgcn_s_wait_tensorcnt(0);
    }

    // stage B tile 32x64 transposed -> Bsh[n][k] (f32 -> f16)
    {
      const int kk = tid >> 3;
      const int nb = (tid & 7) * 8;
      const float* src = w + (size_t)(k0 + kk) * QKVD + n0 + nb;
#pragma unroll
      for (int j = 0; j < 8; j += 4) {
        float4 f = *(const float4*)(src + j);
        Bsh[(nb + j + 0) * 40 + kk] = (_Float16)f.x;
        Bsh[(nb + j + 1) * 40 + kk] = (_Float16)f.y;
        Bsh[(nb + j + 2) * 40 + kk] = (_Float16)f.z;
        Bsh[(nb + j + 3) * 40 + kk] = (_Float16)f.w;
      }
    }
    __syncthreads();

    // A fragment: lane holds row m = lane%16; k = 8g..8g+7, 16+8g..16+8g+7
    // (read f32 from the TDM tile, convert to f16 in registers)
    U16 af;
    {
      const float* ar = &Ashf[(wv * 16 + ml) * 32];
      float4 a0 = *(const float4*)(ar + 8 * g);
      float4 a1 = *(const float4*)(ar + 8 * g + 4);
      float4 a2 = *(const float4*)(ar + 16 + 8 * g);
      float4 a3 = *(const float4*)(ar + 16 + 8 * g + 4);
      const float av[16] = { a0.x, a0.y, a0.z, a0.w, a1.x, a1.y, a1.z, a1.w,
                             a2.x, a2.y, a2.z, a2.w, a3.x, a3.y, a3.z, a3.w };
#pragma unroll
      for (int i = 0; i < 16; ++i) af.v[i] = (_Float16)av[i];
    }
#pragma unroll
    for (int ns = 0; ns < 4; ++ns) {
      U16 bf;
      const _Float16* bp = &Bsh[(ns * 16 + ml) * 40 + 8 * g];
      bf.h8[0] = *(const v8h*)(bp);
      bf.h8[1] = *(const v8h*)(bp + 16);
      acc[ns] = wmma_f16(af.v, bf.v, acc[ns]);
    }
  }

  // epilogue: add bias, scatter as f16 into per-head layouts
#pragma unroll
  for (int ns = 0; ns < 4; ++ns) {
    const int col  = n0 + ns * 16 + ml;
    const int sel  = col / OUTD;            // uniform within a block (64 | 384)
    const int head = (col % OUTD) / HD;
    const int d    = col % HD;
    _Float16* dst = (sel == 0) ? qraw : (sel == 1) ? kraw : vraw;
    const float bv = bias[col];
#pragma unroll
    for (int e = 0; e < 8; ++e) {
      const int r = m0 + wv * 16 + e + 8 * g;
      const int b = r / NTOK;
      const int n = r % NTOK;
      dst[(((size_t)(b * NH + head)) * NTOK + n) * HD + d] =
          (_Float16)(acc[ns][e] + bv);
    }
  }
}

// ---------------------------------------------------------------------------
// Kernel 2: depthwise 3x3 stride-2 pool + LayerNorm
// z==0 -> q (f32 + f16*SCALE), z==1 -> k (f16), z==2 -> v transposed (f16)
// ---------------------------------------------------------------------------
__global__ __launch_bounds__(96) void pool_ln_kernel(
    const _Float16* __restrict__ qraw, const _Float16* __restrict__ kraw,
    const _Float16* __restrict__ vraw,
    const float* __restrict__ wq, const float* __restrict__ wk,
    const float* __restrict__ wvp,
    const float* __restrict__ gq, const float* __restrict__ bq,
    const float* __restrict__ gk, const float* __restrict__ bk,
    const float* __restrict__ gv, const float* __restrict__ bv,
    float* __restrict__ qf32, _Float16* __restrict__ qf16,
    _Float16* __restrict__ kf16, _Float16* __restrict__ vT)
{
  const int d  = threadIdx.x;
  const int p  = blockIdx.x;
  const int bh = blockIdx.y;
  const int z  = blockIdx.z;
  const int oy = p / OHW, ox = p % OHW;

  const _Float16* raw = (z == 0) ? qraw : (z == 1) ? kraw : vraw;
  const float*    wgt = (z == 0) ? wq   : (z == 1) ? wk   : wvp;
  const float*    gg  = (z == 0) ? gq   : (z == 1) ? gk   : gv;
  const float*    bb  = (z == 0) ? bq   : (z == 1) ? bk   : bv;

  const _Float16* base = raw + (size_t)bh * NTOK * HD;
  float acc = 0.0f;
#pragma unroll
  for (int r = 0; r < 3; ++r) {
    const int iy = oy * 2 - 1 + r;
    if (iy < 0 || iy >= H_IN) continue;
#pragma unroll
    for (int c = 0; c < 3; ++c) {
      const int ix = ox * 2 - 1 + c;
      if (ix < 0 || ix >= W_IN) continue;
      acc += (float)base[(size_t)(iy * W_IN + ix) * HD + d] * wgt[d * 9 + r * 3 + c];
    }
  }

  __shared__ float rs[96], rq[96];
  __shared__ float smv[2];
  rs[d] = acc;
  rq[d] = acc * acc;
  __syncthreads();
  if (d < 32) {
    float s  = rs[d] + rs[d + 32] + rs[d + 64];
    float q2 = rq[d] + rq[d + 32] + rq[d + 64];
#pragma unroll
    for (int m = 16; m; m >>= 1) {
      s  += __shfl_xor(s, m, 32);
      q2 += __shfl_xor(q2, m, 32);
    }
    if (d == 0) {
      const float mean = s / 96.0f;
      smv[0] = mean;
      smv[1] = q2 / 96.0f - mean * mean;
    }
  }
  __syncthreads();
  const float mean = smv[0], var = smv[1];
  const float y = (acc - mean) * rsqrtf(var + 1e-6f) * gg[d] + bb[d];

  if (z == 0) {
    qf32[((size_t)bh * NPOOL + p) * HD + d] = y;
    qf16[((size_t)bh * NPAD + p) * HD + d]  = (_Float16)(y * SCALE_F);
  } else if (z == 1) {
    kf16[((size_t)bh * NPAD + p) * HD + d]  = (_Float16)y;
  } else {
    vT[((size_t)bh * HD + d) * NPAD + p]    = (_Float16)y;
  }
}

// zero the padded key columns of vT so 0*garbage can never be NaN
__global__ void zero_vt_pad_kernel(_Float16* __restrict__ vT) {
  const int i = blockIdx.x * blockDim.x + threadIdx.x;   // 64*96*16 pads
  if (i >= NBH * HD * (NPAD - NPOOL)) return;
  const int row = i / (NPAD - NPOOL);
  const int c   = i % (NPAD - NPOOL);
  vT[(size_t)row * NPAD + NPOOL + c] = (_Float16)0.0f;
}

// ---------------------------------------------------------------------------
// Kernel 3: decomposed rel-pos:  eh[bh,q,28], ew[bh,q,28]
// ---------------------------------------------------------------------------
__global__ __launch_bounds__(64) void relpos_kernel(
    const float* __restrict__ qf32, const float* __restrict__ rel_h,
    const float* __restrict__ rel_w,
    float* __restrict__ ehb, float* __restrict__ ewb)
{
  const int j  = threadIdx.x;
  const int p  = blockIdx.x;
  const int bh = blockIdx.y;
  __shared__ float qrow[HD];
  const float* q = qf32 + ((size_t)bh * NPOOL + p) * HD;
  for (int c = j; c < HD; c += 64) qrow[c] = q[c];
  __syncthreads();
  if (j >= 56) return;

  const int qy = p / OHW, qx = p % OHW;
  const float* tab;
  float* dst;
  int idx, kout;
  if (j < 28) { kout = j;      idx = qy - j + 27;          tab = rel_h; dst = ehb; }
  else        { kout = j - 28; idx = qx - (j - 28) + 27;   tab = rel_w; dst = ewb; }
  const float* trow = tab + (size_t)idx * HD;
  float s = 0.0f;
#pragma unroll 4
  for (int c = 0; c < HD; ++c) s += qrow[c] * trow[c];
  dst[((size_t)bh * NPOOL + p) * OHW + kout] = s;
}

// ---------------------------------------------------------------------------
// Kernel 4: flash attention. 1 wave per 16-query tile, 32-key tiles.
// ---------------------------------------------------------------------------
__global__ __launch_bounds__(128) void attn_kernel(
    const _Float16* __restrict__ qf16, const _Float16* __restrict__ kf16,
    const _Float16* __restrict__ vT,   const float* __restrict__ qf32,
    const float* __restrict__ ehb,     const float* __restrict__ ewb,
    float* __restrict__ out)
{
  __shared__ __attribute__((aligned(16))) _Float16 Pl[4][16 * 32];

  const int lane = threadIdx.x & 31;
  const int wv   = threadIdx.x >> 5;
  const int qt   = blockIdx.x * 4 + wv;
  if (qt >= 49) return;                         // wave-uniform exit
  const int bh = blockIdx.y;
  const int b  = bh >> 2, h = bh & 3;
  const int qbase = qt * 16;
  const int g  = lane >> 4, nl = lane & 15;

  // Q fragments (SCALE pre-folded), resident for the whole key loop
  U16 qfr[3];
  {
    const _Float16* qp = qf16 + ((size_t)bh * NPAD + qbase + nl) * HD;
#pragma unroll
    for (int kc = 0; kc < 3; ++kc) {
      qfr[kc].h8[0] = *(const v8h*)(qp + kc * 32 + 8 * g);
      qfr[kc].h8[1] = *(const v8h*)(qp + kc * 32 + 16 + 8 * g);
    }
  }

  float rmax[8], rsum[8];
  v8f O[6];
#pragma unroll
  for (int e = 0; e < 8; ++e) { rmax[e] = -1e30f; rsum[e] = 0.0f; }
#pragma unroll
  for (int t = 0; t < 6; ++t) O[t] = v8f_zero();

  const float* ehrow = ehb + (size_t)bh * NPOOL * OHW;
  const float* ewrow = ewb + (size_t)bh * NPOOL * OHW;
  _Float16* P = Pl[wv];

  for (int kt = 0; kt < 25; ++kt) {
    const int ks0 = kt * 32;
    if (kt + 1 < 25)
      __builtin_prefetch(kf16 + ((size_t)bh * NPAD + ks0 + 32 + nl) * HD, 0, 1);

    v8f S[2];
#pragma unroll
    for (int sub = 0; sub < 2; ++sub) {
      const int ks = ks0 + sub * 16;
      if (ks < NPOOL) {                         // uniform branch, EXEC stays full
        v8f s = v8f_zero();
        const _Float16* kp = kf16 + ((size_t)bh * NPAD + ks + nl) * HD;
#pragma unroll
        for (int kc = 0; kc < 3; ++kc) {
          U16 kf;
          kf.h8[0] = *(const v8h*)(kp + kc * 32 + 8 * g);
          kf.h8[1] = *(const v8h*)(kp + kc * 32 + 16 + 8 * g);
          s = wmma_f16(qfr[kc].v, kf.v, s);
        }
        const int n  = ks + nl;
        const int kh = n / OHW, kw = n % OHW;
#pragma unroll
        for (int e = 0; e < 8; ++e) {
          const int qi = qbase + e + 8 * g;
          s[e] += ehrow[(size_t)qi * OHW + kh] + ewrow[(size_t)qi * OHW + kw];
        }
        S[sub] = s;
      } else {
#pragma unroll
        for (int e = 0; e < 8; ++e) S[sub][e] = -1e30f;
      }
    }

    // online softmax (row m = e + 8g lives across the 16-lane n-group)
#pragma unroll
    for (int e = 0; e < 8; ++e) {
      float mx = fmaxf(S[0][e], S[1][e]);
#pragma unroll
      for (int msk = 8; msk; msk >>= 1) mx = fmaxf(mx, __shfl_xor(mx, msk, 32));
      const float mnew = fmaxf(rmax[e], mx);
      const float p0 = __expf(S[0][e] - mnew);
      const float p1 = __expf(S[1][e] - mnew);
      float sm = p0 + p1;
#pragma unroll
      for (int msk = 8; msk; msk >>= 1) sm += __shfl_xor(sm, msk, 32);
      const float corr = __expf(rmax[e] - mnew);
      rsum[e] = rsum[e] * corr + sm;
      rmax[e] = mnew;
#pragma unroll
      for (int t = 0; t < 6; ++t) O[t][e] *= corr;
      const int m = e + 8 * g;                  // D-layout -> LDS row-major P
      P[m * 32 + nl]      = (_Float16)p0;
      P[m * 32 + 16 + nl] = (_Float16)p1;
    }
    asm volatile("s_wait_dscnt 0" ::: "memory"); // CDNA5 split-counter wait

    // reload P in WMMA A layout
    U16 pf;
    pf.h8[0] = *(const v8h*)&P[nl * 32 + 8 * g];
    pf.h8[1] = *(const v8h*)&P[nl * 32 + 16 + 8 * g];
#pragma unroll
    for (int t = 0; t < 6; ++t) {
      const _Float16* vp = vT + ((size_t)bh * HD + t * 16 + nl) * NPAD + ks0 + 8 * g;
      U16 vf;
      vf.h8[0] = *(const v8h*)(vp);
      vf.h8[1] = *(const v8h*)(vp + 16);
      O[t] = wmma_f16(pf.v, vf.v, O[t]);
    }
  }

  // epilogue: normalize, residual (+q), transpose to [b, tok, h*96+d]
#pragma unroll
  for (int t = 0; t < 6; ++t) {
    const int d = t * 16 + nl;
#pragma unroll
    for (int e = 0; e < 8; ++e) {
      const int qi = qbase + e + 8 * g;
      const float val = O[t][e] / rsum[e] +
                        qf32[((size_t)bh * NPOOL + qi) * HD + d];
      out[((size_t)b * NPOOL + qi) * OUTD + h * HD + d] = val;
    }
  }
}

// ---------------------------------------------------------------------------
extern "C" void kernel_launch(void* const* d_in, const int* in_sizes, int n_in,
                              void* d_out, int out_size, void* d_ws, size_t ws_size,
                              hipStream_t stream) {
  (void)in_sizes; (void)n_in; (void)out_size; (void)ws_size;
  const float* hs    = (const float*)d_in[0];
  const float* qkv_w = (const float*)d_in[1];
  const float* qkv_b = (const float*)d_in[2];
  const float* wq    = (const float*)d_in[3];
  const float* wk    = (const float*)d_in[4];
  const float* wvp   = (const float*)d_in[5];
  const float* gq    = (const float*)d_in[6];
  const float* bq    = (const float*)d_in[7];
  const float* gk    = (const float*)d_in[8];
  const float* bk    = (const float*)d_in[9];
  const float* gv    = (const float*)d_in[10];
  const float* bv    = (const float*)d_in[11];
  const float* rel_h = (const float*)d_in[12];
  const float* rel_w = (const float*)d_in[13];
  float* out = (float*)d_out;

  char* ws = (char*)d_ws;
  size_t off = 0;
  auto take = [&](size_t bytes) -> char* {
    off = (off + 255) & ~(size_t)255;
    char* p = ws + off;
    off += bytes;
    return p;
  };

  _Float16* qraw = (_Float16*)take((size_t)NBH * NTOK * HD * 2);
  _Float16* kraw = (_Float16*)take((size_t)NBH * NTOK * HD * 2);
  _Float16* vraw = (_Float16*)take((size_t)NBH * NTOK * HD * 2);
  float*    qf32 = (float*)   take((size_t)NBH * NPOOL * HD * 4);
  _Float16* qf16 = (_Float16*)take((size_t)NBH * NPAD * HD * 2);
  _Float16* kf16 = (_Float16*)take((size_t)NBH * NPAD * HD * 2);
  _Float16* vT   = (_Float16*)take((size_t)NBH * HD * NPAD * 2);
  float*    ehb  = (float*)   take((size_t)NBH * NPOOL * OHW * 4);
  float*    ewb  = (float*)   take((size_t)NBH * NPOOL * OHW * 4);

  // 1) QKV projection (WMMA f16, A tile via Tensor Data Mover)
  dim3 g1((NB * NTOK) / 128, QKVD / 64);
  qkv_gemm_kernel<<<g1, 256, 0, stream>>>(hs, qkv_w, qkv_b, qraw, kraw, vraw);

  // 2) pool + layernorm (q/k/v via blockIdx.z)
  dim3 g2(NPOOL, NBH, 3);
  pool_ln_kernel<<<g2, 96, 0, stream>>>(qraw, kraw, vraw, wq, wk, wvp,
                                        gq, bq, gk, bk, gv, bv,
                                        qf32, qf16, kf16, vT);

  // 2b) zero vT padding columns
  const int npad_elems = NBH * HD * (NPAD - NPOOL);
  zero_vt_pad_kernel<<<(npad_elems + 255) / 256, 256, 0, stream>>>(vT);

  // 3) rel-pos tables
  dim3 g3(NPOOL, NBH);
  relpos_kernel<<<g3, 64, 0, stream>>>(qf32, rel_h, rel_w, ehb, ewb);

  // 4) flash attention + residual + output transpose (WMMA f16)
  dim3 g4(13, NBH);   // 13*4 waves >= 49 q-tiles
  attn_kernel<<<g4, 128, 0, stream>>>(qf16, kf16, vT, qf32, ehb, ewb, out);
}
